// CrossAttention_54013508715099
// MI455X (gfx1250) — matrix operand order repaired and checked by reading, not compiled, not run on hardware
//
#include <hip/hip_runtime.h>

// ---------------------------------------------------------------------------
// CrossAttention for MI455X (gfx1250, wave32, WMMA bf16 path)
// B=8, LQ=2048, LKV=1024, D=1024, H=16, HD=64
// ---------------------------------------------------------------------------

typedef float  v8f   __attribute__((ext_vector_type(8)));
typedef __bf16 v16bf __attribute__((ext_vector_type(16)));
typedef __bf16 v8bf  __attribute__((ext_vector_type(8)));

#define WMMA_BF16(A, B, C) \
  __builtin_amdgcn_wmma_f32_16x16x32_bf16(false, (A), false, (B), (short)0, (C), false, false)

// A-matrix 16x32 bf16 fragment: lane half hf holds K chunks {8*hf.., 16+8*hf..}
__device__ __forceinline__ v16bf load_a_frag(const __bf16* p, int hf) {
  v8bf lo = *(const v8bf*)(p + 8 * hf);
  v8bf hi = *(const v8bf*)(p + 16 + 8 * hf);
  return __builtin_shufflevector(lo, hi, 0, 1, 2, 3, 4, 5, 6, 7, 8, 9, 10, 11, 12, 13, 14, 15);
}
// B-matrix 32x16 bf16 fragment: lane half hf holds K chunks {8*hf.., 16+8*hf..}
__device__ __forceinline__ v16bf load_b_frag(const __bf16* p, int hf) {
  v8bf lo = *(const v8bf*)(p + 8 * hf);
  v8bf hi = *(const v8bf*)(p + 16 + 8 * hf);
  return __builtin_shufflevector(lo, hi, 0, 1, 2, 3, 4, 5, 6, 7, 8, 9, 10, 11, 12, 13, 14, 15);
}

// ---------------------------------------------------------------------------
// fp32 -> bf16 elementwise convert
// ---------------------------------------------------------------------------
__global__ void cvt_f32_bf16(const float* __restrict__ s, __bf16* __restrict__ d, int n) {
  int i = blockIdx.x * blockDim.x + threadIdx.x;
  if (i < n) d[i] = (__bf16)s[i];
}

// ---------------------------------------------------------------------------
// per-batch "has any valid token" reduction
// ---------------------------------------------------------------------------
__global__ void mask_any_kernel(const int* __restrict__ m, int* __restrict__ out) {
  __shared__ int red;
  if (threadIdx.x == 0) red = 0;
  __syncthreads();
  int b = blockIdx.x;
  int v = 0;
  for (int i = threadIdx.x; i < 1024; i += blockDim.x) v |= m[b * 1024 + i];
  if (v) atomicOr(&red, 1);
  __syncthreads();
  if (threadIdx.x == 0) out[b] = red;
}

// ---------------------------------------------------------------------------
// Projection GEMM: Y[M,1024] = A[M,1024] @ W[1024,1024]^T  (bf16 in, f32 acc)
// Each wave: 32(M) x 64(N) tile (2 A-frags reuse 4 B-frags -> 8 WMMA / k-step).
// N-tile == one head for MODE 0/2.
// MODE 0: RMS-norm epilogue over the 64-wide head, store bf16 [B,H,L,64]
// MODE 2: store bf16 transposed [B,H,64,L]   (for V)
// MODE 3: += bias, store fp32 [M,1024]       (final output projection)
// ---------------------------------------------------------------------------
template <int MODE>
__global__ __launch_bounds__(256) void proj_kernel(
    const __bf16* __restrict__ A, const __bf16* __restrict__ W,
    const float* __restrict__ aux, void* __restrict__ outp, int lsh) {
  const int lane = threadIdx.x & 31;
  const int hf = lane >> 4;
  const int l16 = lane & 15;
  const int wid = blockIdx.x * 8 + (threadIdx.x >> 5);
  const int m0 = (wid >> 4) * 32;
  const int tn = wid & 15;  // head index for MODE 0/2
  const int n0 = tn * 64;

  v8f z = {0.f, 0.f, 0.f, 0.f, 0.f, 0.f, 0.f, 0.f};
  v8f acc[2][4] = {{z, z, z, z}, {z, z, z, z}};
  const __bf16* arow0 = A + (size_t)(m0 + l16) * 1024;
  const __bf16* arow1 = arow0 + (size_t)16 * 1024;
  const __bf16* wrow = W + (size_t)(n0 + l16) * 1024;

  for (int kk = 0; kk < 1024; kk += 32) {
    v16bf af0 = load_a_frag(arow0 + kk, hf);
    v16bf af1 = load_a_frag(arow1 + kk, hf);
#pragma unroll
    for (int j = 0; j < 4; ++j) {
      v16bf bf = load_b_frag(wrow + (size_t)j * 16 * 1024 + kk, hf);
      acc[0][j] = WMMA_BF16(af0, bf, acc[0][j]);
      acc[1][j] = WMMA_BF16(af1, bf, acc[1][j]);
    }
  }

  const int Lmask = (1 << lsh) - 1;
  if (MODE == 0) {
    __bf16* out = (__bf16*)outp;
#pragma unroll
    for (int i = 0; i < 2; ++i) {
      // RMS norm across the 64 columns (one head) per output row
      float rinv[8];
#pragma unroll
      for (int r = 0; r < 8; ++r) {
        float s = 0.f;
#pragma unroll
        for (int j = 0; j < 4; ++j) { float v = acc[i][j][r]; s += v * v; }
        s += __shfl_xor(s, 1, 32);
        s += __shfl_xor(s, 2, 32);
        s += __shfl_xor(s, 4, 32);
        s += __shfl_xor(s, 8, 32);
        rinv[r] = rsqrtf(s * (1.0f / 64.0f) + 1e-6f);
      }
#pragma unroll
      for (int j = 0; j < 4; ++j) {
        int d = j * 16 + l16;
        float wd = aux[d];
#pragma unroll
        for (int r = 0; r < 8; ++r) {
          int gm = m0 + i * 16 + r + 8 * hf;
          int b = gm >> lsh, lr = gm & Lmask;
          size_t idx = (((size_t)(b * 16 + tn) << lsh) + lr) * 64 + d;
          out[idx] = (__bf16)(acc[i][j][r] * rinv[r] * wd);
        }
      }
    }
  } else if (MODE == 2) {
    __bf16* out = (__bf16*)outp;
#pragma unroll
    for (int i = 0; i < 2; ++i)
#pragma unroll
      for (int j = 0; j < 4; ++j) {
        int d = j * 16 + l16;
#pragma unroll
        for (int r = 0; r < 8; ++r) {
          int gm = m0 + i * 16 + r + 8 * hf;
          int b = gm >> lsh, lr = gm & Lmask;
          size_t idx = (((size_t)(b * 16 + tn) * 64 + d) << lsh) + lr;
          out[idx] = (__bf16)acc[i][j][r];
        }
      }
  } else {  // MODE == 3
    float* out = (float*)outp;
#pragma unroll
    for (int i = 0; i < 2; ++i)
#pragma unroll
      for (int j = 0; j < 4; ++j) {
        int n = n0 + j * 16 + l16;
        float bias = aux[n];
#pragma unroll
        for (int r = 0; r < 8; ++r) {
          int gm = m0 + i * 16 + r + 8 * hf;
          out[(size_t)gm * 1024 + n] = acc[i][j][r] + bias;
        }
      }
  }
}

// ---------------------------------------------------------------------------
// Flash-style attention. One wave = one (b,h) x 16-query tile.
// q: bf16 [B,H,LQ,64], k: bf16 [B,H,LKV,64], vt: bf16 [B,H,64,LKV]
// out: bf16 [B,LQ,1024]  (head-interleaved, feeds final projection)
// ---------------------------------------------------------------------------
__global__ __launch_bounds__(256) void attn_kernel(
    const __bf16* __restrict__ q, const __bf16* __restrict__ k,
    const __bf16* __restrict__ vt, const int* __restrict__ cmask,
    const int* __restrict__ has_any, __bf16* __restrict__ outp) {
  __shared__ __align__(32) __bf16 s_p[8][16 * 32];  // P tile per wave
  __shared__ float s_bc[8][16];                     // per-row broadcast
  __shared__ int s_mask[8][32];                     // mask tile per wave

  const int lane = threadIdx.x & 31;
  const int w = threadIdx.x >> 5;
  const int hf = lane >> 4;
  const int l16 = lane & 15;
  const int wid = blockIdx.x * 8 + w;
  const int bh = wid >> 7;  // LQ/16 = 128 q-tiles
  const int qt = wid & 127;
  const int b = bh >> 4;
  const int h = bh & 15;

  const __bf16* qp = q + ((size_t)bh * 2048 + qt * 16) * 64;
  const __bf16* kp = k + (size_t)bh * 1024 * 64;
  const __bf16* vp = vt + (size_t)bh * 64 * 1024;
  const int* mp = cmask + b * 1024;
  const bool anyv = has_any[b] != 0;

  // Q as B-operand (column = query, contiguous over d), loaded once
  v16bf qf0 = load_b_frag(qp + (size_t)l16 * 64 + 0, hf);
  v16bf qf1 = load_b_frag(qp + (size_t)l16 * 64 + 32, hf);

  v8f z = {0.f, 0.f, 0.f, 0.f, 0.f, 0.f, 0.f, 0.f};
  v8f o[4] = {z, z, z, z};
  float mrow = -1e30f, lrow = 0.f;

  for (int kb = 0; kb < 1024; kb += 32) {
    s_mask[w][lane] = mp[kb + lane];

    // S^T[ki, qi] = sum_d K[ki,d] * Q[qi,d]
    v8f sacc[2] = {z, z};
#pragma unroll
    for (int dh = 0; dh < 2; ++dh) {
      v16bf a0 = load_a_frag(kp + (size_t)(kb + l16) * 64 + dh * 32, hf);
      v16bf a1 = load_a_frag(kp + (size_t)(kb + 16 + l16) * 64 + dh * 32, hf);
      v16bf bq = dh ? qf1 : qf0;
      sacc[0] = WMMA_BF16(a0, bq, sacc[0]);
      sacc[1] = WMMA_BF16(a1, bq, sacc[1]);
    }

    asm volatile("s_wait_dscnt 0" ::: "memory");

    // mask + scale; online softmax state per query column (lane pair l16,l16+16)
    float sv[16];
    float tmax = -1e30f;
#pragma unroll
    for (int s = 0; s < 2; ++s)
#pragma unroll
      for (int r = 0; r < 8; ++r) {
        int kloc = s * 16 + r + 8 * hf;
        float x = sacc[s][r] * 0.125f;  // HD^-0.5
        bool valid = (s_mask[w][kloc] != 0) || (!anyv && kb == 0 && kloc == 0);
        x = valid ? x : -1e30f;
        sv[s * 8 + r] = x;
        tmax = fmaxf(tmax, x);
      }
    tmax = fmaxf(tmax, __shfl_xor(tmax, 16, 32));
    float mnew = fmaxf(mrow, tmax);
    float alpha = __expf(mrow - mnew);
    float ps = 0.f;
#pragma unroll
    for (int i = 0; i < 16; ++i) {
      float p = __expf(sv[i] - mnew);
      sv[i] = p;
      ps += p;
    }
    ps += __shfl_xor(ps, 16, 32);
    lrow = lrow * alpha + ps;
    mrow = mnew;

    // broadcast alpha into O-tile row layout; transpose P via LDS
    if (!hf) s_bc[w][l16] = alpha;
#pragma unroll
    for (int s = 0; s < 2; ++s)
#pragma unroll
      for (int r = 0; r < 8; ++r) {
        int kloc = s * 16 + r + 8 * hf;
        s_p[w][l16 * 32 + kloc] = (__bf16)sv[s * 8 + r];
      }
    asm volatile("s_wait_dscnt 0" ::: "memory");

    float al[8];
#pragma unroll
    for (int r = 0; r < 8; ++r) al[r] = s_bc[w][r + 8 * hf];
    v16bf pf = load_a_frag(&s_p[w][l16 * 32], hf);

    // O[qi, d] = O*alpha + P @ V  (V^T: column d contiguous over ki)
#pragma unroll
    for (int t = 0; t < 4; ++t) {
      v16bf vf = load_b_frag(vp + (size_t)(t * 16 + l16) * 1024 + kb, hf);
      v8f oo = o[t];
#pragma unroll
      for (int r = 0; r < 8; ++r) oo[r] *= al[r];
      o[t] = WMMA_BF16(pf, vf, oo);
    }
  }

  // finalize: divide rows by l
  if (!hf) s_bc[w][l16] = lrow;
  asm volatile("s_wait_dscnt 0" ::: "memory");
  float rl[8];
#pragma unroll
  for (int r = 0; r < 8; ++r) rl[r] = 1.0f / s_bc[w][r + 8 * hf];

#pragma unroll
  for (int t = 0; t < 4; ++t) {
    int d = t * 16 + l16;
#pragma unroll
    for (int r = 0; r < 8; ++r) {
      int qi = r + 8 * hf;
      size_t row = (size_t)b * 2048 + qt * 16 + qi;
      outp[row * 1024 + h * 64 + d] = (__bf16)(o[t][r] * rl[r]);
    }
  }
}

// ---------------------------------------------------------------------------
// host launch
// ---------------------------------------------------------------------------
extern "C" void kernel_launch(void* const* d_in, const int* in_sizes, int n_in,
                              void* d_out, int out_size, void* d_ws, size_t ws_size,
                              hipStream_t stream) {
  const float* x = (const float*)d_in[0];
  const float* context = (const float*)d_in[1];
  const int* cmask = (const int*)d_in[2];
  const float* Wq = (const float*)d_in[3];
  const float* Wk = (const float*)d_in[4];
  const float* Wv = (const float*)d_in[5];
  const float* Wp = (const float*)d_in[6];
  const float* bp = (const float*)d_in[7];
  const float* qnw = (const float*)d_in[8];
  const float* knw = (const float*)d_in[9];

  char* ws = (char*)d_ws;
  const size_t SZ_W = (size_t)1024 * 1024 * 2;  // 2MB per bf16 weight
  __bf16* wq_b = (__bf16*)(ws + 0 * SZ_W);
  __bf16* wk_b = (__bf16*)(ws + 1 * SZ_W);
  __bf16* wv_b = (__bf16*)(ws + 2 * SZ_W);
  __bf16* wp_b = (__bf16*)(ws + 3 * SZ_W);
  size_t off = 4 * SZ_W;
  __bf16* x_b = (__bf16*)(ws + off);  off += (size_t)16777216 * 2;  // 32MB
  __bf16* c_b = (__bf16*)(ws + off);  off += (size_t)8388608 * 2;   // 16MB
  __bf16* q_b = (__bf16*)(ws + off);  off += (size_t)16777216 * 2;  // 32MB
  __bf16* k_b = (__bf16*)(ws + off);  off += (size_t)8388608 * 2;   // 16MB
  __bf16* vt_b = (__bf16*)(ws + off); off += (size_t)8388608 * 2;   // 16MB
  int* hany = (int*)(ws + off);       off += 256;
  __bf16* ao_b = x_b;  // x_b is dead after the Q projection; reuse for attn out

  const int nW = 1024 * 1024;
  cvt_f32_bf16<<<(nW + 255) / 256, 256, 0, stream>>>(Wq, wq_b, nW);
  cvt_f32_bf16<<<(nW + 255) / 256, 256, 0, stream>>>(Wk, wk_b, nW);
  cvt_f32_bf16<<<(nW + 255) / 256, 256, 0, stream>>>(Wv, wv_b, nW);
  cvt_f32_bf16<<<(nW + 255) / 256, 256, 0, stream>>>(Wp, wp_b, nW);
  const int nx = 8 * 2048 * 1024;
  const int nc = 8 * 1024 * 1024;
  cvt_f32_bf16<<<(nx + 255) / 256, 256, 0, stream>>>(x, x_b, nx);
  cvt_f32_bf16<<<(nc + 255) / 256, 256, 0, stream>>>(context, c_b, nc);

  // Q projection + q rms-norm: M=16384 rows -> (16384/32)*16 = 8192 waves
  proj_kernel<0><<<1024, 256, 0, stream>>>(x_b, wq_b, qnw, q_b, 11);
  // K projection + k rms-norm: M=8192 -> 4096 waves
  proj_kernel<0><<<512, 256, 0, stream>>>(c_b, wk_b, knw, k_b, 10);
  // V projection with transpose epilogue
  proj_kernel<2><<<512, 256, 0, stream>>>(c_b, wv_b, nullptr, vt_b, 10);

  mask_any_kernel<<<8, 256, 0, stream>>>(cmask, hany);

  // attention: 128 (b,h) * 128 q-tiles = 16384 waves
  attn_kernel<<<2048, 256, 0, stream>>>(q_b, k_b, vt_b, cmask, hany, ao_b);

  // output projection + bias -> fp32 d_out
  proj_kernel<3><<<1024, 256, 0, stream>>>(ao_b, wp_b, bp, (float*)d_out, 11);
}
